// poly_33861522162598
// MI455X (gfx1250) — matrix operand order repaired and checked by reading, not compiled, not run on hardware
//
#include <hip/hip_runtime.h>

// ---- CDNA5 WMMA types -------------------------------------------------------
typedef __attribute__((ext_vector_type(16))) __bf16 v16bf;
typedef __attribute__((ext_vector_type(8)))  float  v8f;

union BFrag {
    v16bf v;
    unsigned short s[16];
    uint4 q[2];
};

// Truncate f32 -> bf16 bits (round-toward-zero); residual f - hi is exact in f32.
__device__ __forceinline__ unsigned short bf_hi_trunc(float f, float* hif) {
    unsigned u = __float_as_uint(f) & 0xFFFF0000u;
    *hif = __uint_as_float(u);
    return (unsigned short)(u >> 16);
}
__device__ __forceinline__ unsigned short bf_trunc(float f) {
    return (unsigned short)(__float_as_uint(f) >> 16);
}

// -----------------------------------------------------------------------------
// Pre-pass: pack W (128x512 f32) into fragment-ordered bf16 hi/lo (256 KB).
// Layout (bf16 units): base = ((o*16 + ks)*2 + half)*32
//   wpk[base + e]      = hi(W[o, K(e)])        e = 0..15
//   wpk[base + 16 + e] = lo(W[o, K(e)])
//   K(e) = ks*32 + (e>=8 ? 16 : 0) + half*8 + (e&7)   (ISA 16-bit A 16x32 layout)
// -----------------------------------------------------------------------------
__global__ __launch_bounds__(256) void pack_w_bf16(const float* __restrict__ w,
                                                   unsigned short* __restrict__ wpk)
{
    const int g    = blockIdx.x * 256 + threadIdx.x;   // 0..65535
    const int e    = g & 15;
    const int half = (g >> 4) & 1;
    const int ks   = (g >> 5) & 15;
    const int o    = g >> 9;
    const int K    = ks * 32 + ((e >> 3) << 4) + half * 8 + (e & 7);
    const float v  = w[o * 512 + K];
    float hif;
    const unsigned short hi = bf_hi_trunc(v, &hif);
    const unsigned short lo = bf_trunc(v - hif);
    const int base = (g >> 4) << 5;                    // ((o*16+ks)*2+half)*32
    wpk[base + e]      = hi;
    wpk[base + 16 + e] = lo;
}

// -----------------------------------------------------------------------------
// Main kernel: each block handles one (b,d,h) output row (N = 32 positions).
// 8 waves = 8 M-tiles; each wave computes a 16x32 tile as two 16x16 WMMA tiles
// sharing the A fragment (6 wmma per k-step, two independent acc chains).
// -----------------------------------------------------------------------------
__global__ __launch_bounds__(256) void conv222_wmma_main(
    const float* __restrict__ x, const unsigned short* __restrict__ wpk,
    float* __restrict__ out)
{
    // Patch tile (K=512, N=32) bf16 hi/lo, layout [n][K], stride 520 (bank-friendly).
    __shared__ __align__(16) unsigned short ldsHi[32 * 520];
    __shared__ __align__(16) unsigned short ldsLo[32 * 520];

    const int tid = threadIdx.x;
    const int bid = blockIdx.x;            // 4096 blocks: [b:2][d:5][h:5]
    const int h  = bid & 31;
    const int d  = (bid >> 5) & 31;
    const int b  = bid >> 10;

    // ---- cooperative gather: x -> LDS (hi/lo bf16). With kg = tid>>5 (0..7),
    // K = kg + 8*i keeps K&7 == kg (so kd,kh,kw constant) and c = i, so the 64
    // addresses are base + i*35937. 32 consecutive lanes read 128 contiguous
    // bytes -> fully coalesced. ----
    {
        const int n  = tid & 31;           // output column 0..31
        const int kg = tid >> 5;           // 0..7
        const int kd = kg >> 2, kh = (kg >> 1) & 1, kw = kg & 1;
        const float* xp = x + (size_t)b * (64u * 35937u)
                            + (size_t)(d + kd) * 1089
                            + (size_t)(h + kh) * 33
                            + (n + kw);
        unsigned short* pHi = &ldsHi[n * 520 + kg];
        unsigned short* pLo = &ldsLo[n * 520 + kg];
        #pragma unroll 8
        for (int i = 0; i < 64; ++i) {     // K = kg + 8*i, c = i
            const float v = xp[(size_t)i * 35937];
            float hif;
            pHi[i * 8] = bf_hi_trunc(v, &hif);
            pLo[i * 8] = bf_trunc(v - hif);
        }
    }
    __syncthreads();

    const int lane  = tid & 31;
    const int mtile = tid >> 5;            // 0..7
    const int n     = lane & 15;
    const int half  = lane >> 4;
    const int o     = mtile * 16 + n;      // A row for this lane (M = lane%16)

    // Packed A fragments: frag(ks) at wpk + o*1024 + ks*64 + half*32 (bf16 units)
    const uint4* pa = (const uint4*)(wpk + o * 1024 + half * 32);
    const int boff0 = n * 520 + half * 8;          // columns 0..15
    const int boff1 = (n + 16) * 520 + half * 8;   // columns 16..31

    v8f acc0 = {}, acc1 = {};

    #pragma unroll 2
    for (int ks = 0; ks < 16; ++ks) {
        // ---- A: four 16B loads, already in fragment element order ----
        const uint4* p = pa + ks * 8;      // ks*64 bf16 = ks*8 uint4
        BFrag ahi, alo;
        ahi.q[0] = p[0];  ahi.q[1] = p[1];
        alo.q[0] = p[2];  alo.q[1] = p[3];

        // ---- B from LDS: K = ks*32 + (e>=8 ? 16 : 0) + half*8 + (e&7) (16B aligned) ----
        const int o0 = boff0 + ks * 32;
        const int o1 = boff1 + ks * 32;
        BFrag b0h, b0l, b1h, b1l;
        b0h.q[0] = *(const uint4*)&ldsHi[o0];
        b0h.q[1] = *(const uint4*)&ldsHi[o0 + 16];
        b0l.q[0] = *(const uint4*)&ldsLo[o0];
        b0l.q[1] = *(const uint4*)&ldsLo[o0 + 16];
        b1h.q[0] = *(const uint4*)&ldsHi[o1];
        b1h.q[1] = *(const uint4*)&ldsHi[o1 + 16];
        b1l.q[0] = *(const uint4*)&ldsLo[o1];
        b1l.q[1] = *(const uint4*)&ldsLo[o1 + 16];

        // 3-product bf16 split per sub-tile; interleave the two acc chains.
        acc0 = __builtin_amdgcn_wmma_f32_16x16x32_bf16(false, ahi.v, false, b0h.v,
                                                       (short)0, acc0, false, false);
        acc1 = __builtin_amdgcn_wmma_f32_16x16x32_bf16(false, ahi.v, false, b1h.v,
                                                       (short)0, acc1, false, false);
        acc0 = __builtin_amdgcn_wmma_f32_16x16x32_bf16(false, ahi.v, false, b0l.v,
                                                       (short)0, acc0, false, false);
        acc1 = __builtin_amdgcn_wmma_f32_16x16x32_bf16(false, ahi.v, false, b1l.v,
                                                       (short)0, acc1, false, false);
        acc0 = __builtin_amdgcn_wmma_f32_16x16x32_bf16(false, alo.v, false, b0h.v,
                                                       (short)0, acc0, false, false);
        acc1 = __builtin_amdgcn_wmma_f32_16x16x32_bf16(false, alo.v, false, b1h.v,
                                                       (short)0, acc1, false, false);
    }

    // ---- epilogue: (y+1)^2 ; C/D layout: VGPR r -> M = r + 8*half, N = lane%16 ----
    float* ob = out + ((((size_t)b * 128 + mtile * 16 + half * 8) * 32 + d) * 32 + h) * 32 + n;
    #pragma unroll
    for (int r = 0; r < 8; ++r) {
        const float v0 = acc0[r] + 1.0f;
        const float v1 = acc1[r] + 1.0f;
        ob[(size_t)r * 32768]      = v0 * v0;   // o stride = 32*32*32
        ob[(size_t)r * 32768 + 16] = v1 * v1;   // columns 16..31
    }
}

// -----------------------------------------------------------------------------
// Fallback (no workspace): fused conversion kernel (N=16 per block).
// -----------------------------------------------------------------------------
__global__ __launch_bounds__(256) void conv222_wmma_fused(
    const float* __restrict__ x, const float* __restrict__ w,
    float* __restrict__ out)
{
    __shared__ __align__(16) unsigned short ldsHi[16 * 520];
    __shared__ __align__(16) unsigned short ldsLo[16 * 520];

    const int tid = threadIdx.x;
    const int bid = blockIdx.x;
    const int w0 = (bid & 1) << 4;
    const int h  = (bid >> 1) & 31;
    const int d  = (bid >> 6) & 31;
    const int b  = bid >> 11;

    {
        const int n  = tid & 15;
        const int kg = tid >> 4;
        const int r  = kg & 7;
        const int kd = r >> 2, kh = (r >> 1) & 1, kw = r & 1;
        const float* xp = x + (size_t)b * (64u * 35937u)
                            + (size_t)(kg >> 3) * 35937
                            + (size_t)(d + kd) * 1089
                            + (size_t)(h + kh) * 33
                            + (w0 + n + kw);
        unsigned short* pHi = &ldsHi[n * 520 + kg];
        unsigned short* pLo = &ldsLo[n * 520 + kg];
        #pragma unroll 8
        for (int i = 0; i < 32; ++i) {
            const float v = xp[(size_t)i * (2 * 35937)];
            float hif;
            pHi[i * 16] = bf_hi_trunc(v, &hif);
            pLo[i * 16] = bf_trunc(v - hif);
        }
    }
    __syncthreads();

    const int lane  = tid & 31;
    const int mtile = tid >> 5;
    const int n     = lane & 15;
    const int half  = lane >> 4;

    const float* wrow = w + (size_t)(mtile * 16 + n) * 512;

    v8f acc = {};

    #pragma unroll 4
    for (int ks = 0; ks < 16; ++ks) {
        const float* pa = wrow + ks * 32 + half * 8;
        const float4 a0 = *(const float4*)(pa);
        const float4 a1 = *(const float4*)(pa + 4);
        const float4 a2 = *(const float4*)(pa + 16);
        const float4 a3 = *(const float4*)(pa + 20);
        const float af[16] = {a0.x,a0.y,a0.z,a0.w, a1.x,a1.y,a1.z,a1.w,
                              a2.x,a2.y,a2.z,a2.w, a3.x,a3.y,a3.z,a3.w};
        BFrag ahi, alo;
        #pragma unroll
        for (int e = 0; e < 16; ++e) {
            float hif;
            ahi.s[e] = bf_hi_trunc(af[e], &hif);
            alo.s[e] = bf_trunc(af[e] - hif);
        }
        const int off = n * 520 + ks * 32 + half * 8;
        BFrag bhi, blo;
        bhi.q[0] = *(const uint4*)&ldsHi[off];
        bhi.q[1] = *(const uint4*)&ldsHi[off + 16];
        blo.q[0] = *(const uint4*)&ldsLo[off];
        blo.q[1] = *(const uint4*)&ldsLo[off + 16];

        acc = __builtin_amdgcn_wmma_f32_16x16x32_bf16(false, ahi.v, false, bhi.v,
                                                      (short)0, acc, false, false);
        acc = __builtin_amdgcn_wmma_f32_16x16x32_bf16(false, ahi.v, false, blo.v,
                                                      (short)0, acc, false, false);
        acc = __builtin_amdgcn_wmma_f32_16x16x32_bf16(false, alo.v, false, bhi.v,
                                                      (short)0, acc, false, false);
    }

    float* ob = out + ((((size_t)b * 128 + mtile * 16 + half * 8) * 32 + d) * 32 + h) * 32
                    + (w0 + n);
    #pragma unroll
    for (int r = 0; r < 8; ++r) {
        const float v = acc[r] + 1.0f;
        ob[(size_t)r * 32768] = v * v;
    }
}

extern "C" void kernel_launch(void* const* d_in, const int* in_sizes, int n_in,
                              void* d_out, int out_size, void* d_ws, size_t ws_size,
                              hipStream_t stream) {
    const float* x = (const float*)d_in[0];   // (4,64,33,33,33) f32
    const float* w = (const float*)d_in[1];   // (128,512) f32
    float* out     = (float*)d_out;           // (4,128,32,32,32) f32
    (void)in_sizes; (void)n_in; (void)out_size;

    const size_t need = 128u * 512u * 2u * sizeof(unsigned short);  // 262144 B
    if (d_ws != nullptr && ws_size >= need) {
        unsigned short* wpk = (unsigned short*)d_ws;
        pack_w_bf16<<<dim3(256), dim3(256), 0, stream>>>(w, wpk);
        conv222_wmma_main<<<dim3(4096), dim3(256), 0, stream>>>(x, wpk, out);
    } else {
        conv222_wmma_fused<<<dim3(8192), dim3(256), 0, stream>>>(x, w, out);
    }
}
